// SparseAxialCausalAttention_64183991272016
// MI455X (gfx1250) — matrix operand order repaired and checked by reading, not compiled, not run on hardware
//
#include <hip/hip_runtime.h>

// ---------------------------------------------------------------------------
// SparseAxialCausalAttention on MI455X (gfx1250), wave32 + WMMA bf16 +
// async global->LDS staging (ASYNCcnt) for the dense GEMMs.
//
// Pipeline (all bf16 WMMA, f32 accumulate):
//   1) convertT: W_qkv -> bf16 transposed [1536][512], W_out -> bf16 [512][512]
//   2) qkv_gemm: 8-wave blocks, 128x64 tile; B panel (32x64 bf16 = 4KB/chunk)
//      double-buffered in LDS via global_load_async_to_lds_b128; x f32->bf16
//      on the fly. Writes Q (scaled dh^-0.5), K in [bh][1296][64] bf16 and
//      V transposed [bh][64][1296] bf16.
//   3) attn_text: per (bh, 16-query tile): S=QK^T via v_wmma_f32_16x16x32_bf16,
//      causal mask, LDS softmax, PV via WMMA -> Attn bf16 [b][1296][512]
//   4) attn_img: axial image attention (text keys + same-column image keys,
//      strict causal on image part), same WMMA structure
//   5) out_gemm: same async-staged GEMM: Attn(bf16) @ WoutT + b_out -> f32 out
//
// Workspace use: ~172 MB (WqkvT 1.5M, WoT 0.5M, Q/K/Vt 3x42.5M, Attn 42.5M).
// ---------------------------------------------------------------------------

typedef __bf16 bf16;
typedef __attribute__((ext_vector_type(16))) __bf16 v16bf;
typedef __attribute__((ext_vector_type(8)))  __bf16 v8bf;
typedef __attribute__((ext_vector_type(8)))  float  v8f;
typedef __attribute__((ext_vector_type(4)))  int    v4i;

#define DIMC     512
#define NQKV     1536
#define NPAD     1296      // 81 * 16, covers N = 1281
#define NREAL    1280      // real rows of x
#define TEXTL    257
#define IMGW     32
#define LPAD     320       // padded score-row length (10 chunks of 32)
#define NEGBIG   (-3.0e38f)
#define BSTRIDE  40        // LDS B panel row stride (bf16 elems), bank spread
#define MT128    ((NPAD + 127) / 128)   // 11 row-blocks of 128

// ---- gfx1250 async global->LDS path (probe-guarded) -----------------------
#if defined(__AMDGCN__) && __has_builtin(__builtin_amdgcn_global_load_async_to_lds_b128)
#define USE_ASYNC_LDS 1
typedef __attribute__((address_space(1))) v4i as1_v4i;   // global int4
typedef __attribute__((address_space(3))) v4i as3_v4i;   // LDS int4
#else
#define USE_ASYNC_LDS 0
#endif

__device__ __forceinline__ void async_wait_le1() {
#if USE_ASYNC_LDS
#if __has_builtin(__builtin_amdgcn_s_wait_asynccnt)
  __builtin_amdgcn_s_wait_asynccnt(1);
#else
  asm volatile("s_wait_asynccnt 1" ::: "memory");
#endif
#endif
}
__device__ __forceinline__ void async_wait_le0() {
#if USE_ASYNC_LDS
#if __has_builtin(__builtin_amdgcn_s_wait_asynccnt)
  __builtin_amdgcn_s_wait_asynccnt(0);
#else
  asm volatile("s_wait_asynccnt 0" ::: "memory");
#endif
#endif
}

// Stage 16B (8 bf16) from global to LDS: async DMA if available, else copy.
__device__ __forceinline__ void stage16(const bf16* g, bf16* l) {
#if USE_ASYNC_LDS
  __builtin_amdgcn_global_load_async_to_lds_b128((as1_v4i*)g, (as3_v4i*)l, 0, 0);
#else
  *(v8bf*)l = *(const v8bf*)g;
#endif
}

__device__ __forceinline__ v8f vzero8f() {
  v8f z = {0.f, 0.f, 0.f, 0.f, 0.f, 0.f, 0.f, 0.f};
  return z;
}

__device__ __forceinline__ v16bf vzero16bf() {
  v16bf r;
#pragma unroll
  for (int e = 0; e < 16; ++e) r[e] = (__bf16)0.f;
  return r;
}

// D = A*B + C, 16x16 tile, K=32, bf16 inputs, f32 accumulate.
__device__ __forceinline__ v8f wmma_bf16(v16bf a, v16bf b, v8f c) {
  return __builtin_amdgcn_wmma_f32_16x16x32_bf16(
      /*neg_a=*/false, a, /*neg_b=*/false, b,
      /*c_mod=*/(short)0, c, /*reuse_a=*/false, /*reuse_b=*/false);
}

// --- fragment loaders (wave32; A/B use this file's contiguous k order) -------------------------
// A 16x32 bf16: lane row = lane&15, kb = (lane>>4)*16,
//   element e -> k = kb + e
// B 32x16 bf16: lane col = lane&15, kb = (lane>>4)*16, element e -> k = kb+e
// C 16x16 f32 : lane col = lane&15, vgpr r -> row = (lane>>4)*8 + r

__device__ __forceinline__ v16bf frag_a_bf16(const bf16* base, int kbA) {
  v8bf lo = *(const v8bf*)(base + 2 * kbA);
  v8bf hi = *(const v8bf*)(base + 2 * kbA + 8);
  v16bf r;
#pragma unroll
  for (int e = 0; e < 8; ++e) { r[e] = lo[e]; r[8 + e] = hi[e]; }
  return r;
}

__device__ __forceinline__ v16bf frag_a_f32(const float* base, int kbA) {
  v16bf r;
#pragma unroll
  for (int e = 0; e < 8; ++e) {
    r[e]     = (__bf16)base[2 * kbA + e];
    r[8 + e] = (__bf16)base[2 * kbA + 8 + e];
  }
  return r;
}

// 16 consecutive bf16 as two 16B loads (valid for global and LDS sources).
__device__ __forceinline__ v16bf frag_b(const bf16* p) {
  v8bf lo = *(const v8bf*)p;
  v8bf hi = *(const v8bf*)(p + 8);
  v16bf r;
#pragma unroll
  for (int e = 0; e < 8; ++e) { r[e] = lo[e]; r[8 + e] = hi[e]; }
  return r;
}

// ---------------------------------------------------------------------------
// Kernel 1: f32 -> bf16 transposed weight conversion. W[K][N] -> WT[N][K]
// ---------------------------------------------------------------------------
__global__ void convertT_kernel(const float* __restrict__ W, bf16* __restrict__ WT,
                                int K, int Nc) {
  int idx = blockIdx.x * blockDim.x + threadIdx.x;
  if (idx < K * Nc) {
    int k = idx / Nc, n = idx % Nc;
    WT[(size_t)n * K + k] = (bf16)W[idx];
  }
}

// ---------------------------------------------------------------------------
// Kernel 2: QKV projection. 8 waves/block -> 128(M) x 64(Nout) tile.
// B panel (32x64 bf16 per K-chunk) double-buffered in LDS via async DMA:
// 256 lanes x one b128 each = 4KB per stage.
// grid = (24 col-tiles, 11 row-tiles, 32 batches), block = 256.
// ---------------------------------------------------------------------------
__global__ __launch_bounds__(256)
void qkv_gemm_kernel(const float* __restrict__ x, const bf16* __restrict__ WqkvT,
                     bf16* __restrict__ Qd, bf16* __restrict__ Kd,
                     bf16* __restrict__ Vtd) {
  __shared__ __align__(16) bf16 ldsB[2][64 * BSTRIDE];
  const int tid  = threadIdx.x;
  const int wave = tid >> 5;
  const int lane = tid & 31;
  const int b   = blockIdx.z;
  const int mt  = blockIdx.y;
  const int nt  = blockIdx.x;
  const int row0 = mt * 128 + wave * 16;
  const int col0 = nt * 64;
  const int arow = lane & 15;
  const int kbA  = (lane >> 4) * 8;
  const int bcol = lane & 15;
  const int kbB  = (lane >> 4) * 16;
  const int crb  = (lane >> 4) * 8;

  // staging map: thread -> (col 0..63, 8-elem segment 0..3)
  const int scol = tid >> 2;
  const int sseg = (tid & 3) * 8;
  const bf16* gB = WqkvT + (size_t)(col0 + scol) * DIMC + sseg;
  bf16* lB0 = &ldsB[0][scol * BSTRIDE + sseg];
  bf16* lB1 = &ldsB[1][scol * BSTRIDE + sseg];

  const int grow = row0 + arow;
  const bool azero = (grow >= NREAL);
  const float* arowptr = x + ((size_t)b * NREAL + (azero ? 0 : grow)) * DIMC;
  const v16bf azf = vzero16bf();

  v8f acc[4] = {vzero8f(), vzero8f(), vzero8f(), vzero8f()};

  stage16(gB, lB0);                           // prologue: K-chunk 0 -> buf 0
  for (int i = 0; i < DIMC / 32; ++i) {
    const int kc = i * 32;
    if (i + 1 < DIMC / 32) {                  // next K-chunk -> other buffer
      stage16(gB + kc + 32, (i & 1) ? lB0 : lB1);
      async_wait_le1();
    } else {
      async_wait_le0();
    }
    __syncthreads();
    if (!azero && kc + 32 < DIMC) __builtin_prefetch(arowptr + kc + 32, 0, 1);
    const v16bf a = azero ? azf : frag_a_f32(arowptr + kc, kbA);
    const bf16* Bbase = &ldsB[i & 1][0];
#pragma unroll
    for (int s = 0; s < 4; ++s) {
      const v16bf bb = frag_b(Bbase + (s * 16 + bcol) * BSTRIDE + kbB);
      acc[s] = wmma_bf16(a, bb, acc[s]);
    }
    __syncthreads();                          // all waves done with buf i&1
  }

  const int which = col0 / DIMC;              // 0:q 1:k 2:v
  const int cbase = col0 % DIMC;
#pragma unroll
  for (int s = 0; s < 4; ++s) {
    const int cc = cbase + s * 16 + bcol;     // 0..511
    const int h = cc >> 6, d = cc & 63;
    const size_t bh = (size_t)b * 8 + h;
#pragma unroll
    for (int r = 0; r < 8; ++r) {
      const int pos = row0 + crb + r;
      if (pos < NPAD) {
        const float val = acc[s][r];
        if (which == 0)      Qd[(bh * NPAD + pos) * 64 + d]  = (bf16)(val * 0.125f);
        else if (which == 1) Kd[(bh * NPAD + pos) * 64 + d]  = (bf16)val;
        else                 Vtd[(bh * 64 + d) * NPAD + pos] = (bf16)val;
      }
    }
  }
}

// ---------------------------------------------------------------------------
// Kernel 3: text self-attention (causal). grid = (17 q-tiles, 256 bh), block=32.
// ---------------------------------------------------------------------------
__global__ __launch_bounds__(32)
void attn_text_kernel(const bf16* __restrict__ Q, const bf16* __restrict__ K,
                      const bf16* __restrict__ Vt, bf16* __restrict__ Attn) {
  __shared__ float sc[16][LPAD];
  __shared__ bf16  pr[16][LPAD];
  const int lane = threadIdx.x;
  const int bh = blockIdx.y;
  const int qt = blockIdx.x;
  const int b = bh >> 3, h = bh & 7;
  const int arow = lane & 15;
  const int kbA  = (lane >> 4) * 8;
  const int bcol = lane & 15;
  const int kbB  = (lane >> 4) * 16;
  const int crb  = (lane >> 4) * 8;

  const bf16* Qb = Q  + (size_t)bh * NPAD * 64;
  const bf16* Kb = K  + (size_t)bh * NPAD * 64;
  const bf16* Vb = Vt + (size_t)bh * 64 * NPAD;
  const int q0 = qt * 16;

  // ---- S = Q K^T, causal-masked, into LDS ----
  for (int jt = 0; jt < 17; ++jt) {
    const int j0 = jt * 16;
    v8f c = vzero8f();
#pragma unroll
    for (int kc = 0; kc < 64; kc += 32) {
      const v16bf a  = frag_a_bf16(Qb + (size_t)(q0 + arow) * 64 + kc, kbA);
      const v16bf bb = frag_b(Kb + (size_t)(j0 + bcol) * 64 + kc + kbB);
      c = wmma_bf16(a, bb, c);
    }
    const int j = j0 + bcol;
#pragma unroll
    for (int r = 0; r < 8; ++r) {
      const int qq = q0 + crb + r;
      float s = c[r];
      if (j > qq || j >= TEXTL) s = NEGBIG;
      sc[crb + r][j] = s;
    }
  }
  __syncthreads();

  // ---- row softmax (lanes 0..15 each own one row) ----
  if (lane < 16) {
    float mx = NEGBIG;
    for (int cI = 0; cI < 272; ++cI) mx = fmaxf(mx, sc[lane][cI]);
    float sum = 0.f;
    for (int cI = 0; cI < 272; ++cI) {
      const float e = __expf(sc[lane][cI] - mx);
      sum += e;
      sc[lane][cI] = e;
    }
    const float inv = 1.f / sum;
    for (int cI = 0; cI < 272; ++cI) pr[lane][cI] = (bf16)(sc[lane][cI] * inv);
    for (int cI = 272; cI < 288; ++cI) pr[lane][cI] = (bf16)0.f;
  }
  __syncthreads();

  // ---- O = P V ----
  v8f o[4] = {vzero8f(), vzero8f(), vzero8f(), vzero8f()};
  for (int ct = 0; ct < 9; ++ct) {
    const int c0 = ct * 32;
    const v16bf a = frag_a_bf16(&pr[arow][c0], kbA);
#pragma unroll
    for (int s4 = 0; s4 < 4; ++s4) {
      const int d = s4 * 16 + bcol;
      const v16bf bb = frag_b(Vb + (size_t)d * NPAD + c0 + kbB);
      o[s4] = wmma_bf16(a, bb, o[s4]);
    }
  }
#pragma unroll
  for (int s4 = 0; s4 < 4; ++s4) {
    const int d = s4 * 16 + bcol;
#pragma unroll
    for (int r = 0; r < 8; ++r) {
      const int qq = q0 + crb + r;
      if (qq < TEXTL)
        Attn[((size_t)b * NPAD + qq) * 512 + h * 64 + d] = (bf16)o[s4][r];
    }
  }
}

// ---------------------------------------------------------------------------
// Kernel 4: axial image attention. Queries (xg, i), keys = text (mask) +
// image rows j<i in the same column xg. grid = (2 q-tiles, 32 cols, 256 bh).
// ---------------------------------------------------------------------------
__global__ __launch_bounds__(32)
void attn_img_kernel(const bf16* __restrict__ Q, const bf16* __restrict__ K,
                     const bf16* __restrict__ Vt, const unsigned char* __restrict__ mask,
                     bf16* __restrict__ Attn) {
  __shared__ float sc[16][LPAD];
  __shared__ bf16  pr[16][LPAD];
  const int lane = threadIdx.x;
  const int bh = blockIdx.z;
  const int xg = blockIdx.y;
  const int qt = blockIdx.x;
  const int b = bh >> 3, h = bh & 7;
  const int arow = lane & 15;
  const int kbA  = (lane >> 4) * 8;
  const int bcol = lane & 15;
  const int kbB  = (lane >> 4) * 16;
  const int crb  = (lane >> 4) * 8;

  const bf16* Qb = Q  + (size_t)bh * NPAD * 64;
  const bf16* Kb = K  + (size_t)bh * NPAD * 64;
  const bf16* Vb = Vt + (size_t)bh * 64 * NPAD;

  const int ibase = TEXTL + xg * IMGW;   // sequence position of image row 0, col xg
  const int i0    = qt * 16;             // query row index within the column
  const int qpos0 = ibase + i0;          // sequence position of first query

  // ---- text-key scores (cols 0..271) ----
  for (int jt = 0; jt < 17; ++jt) {
    const int j0 = jt * 16;
    v8f c = vzero8f();
#pragma unroll
    for (int kc = 0; kc < 64; kc += 32) {
      const v16bf a  = frag_a_bf16(Qb + (size_t)(qpos0 + arow) * 64 + kc, kbA);
      const v16bf bb = frag_b(Kb + (size_t)(j0 + bcol) * 64 + kc + kbB);
      c = wmma_bf16(a, bb, c);
    }
    const int j = j0 + bcol;
    bool mj = (j >= TEXTL);
    if (!mj) mj = (mask[b * TEXTL + j] == 0);
#pragma unroll
    for (int r = 0; r < 8; ++r)
      sc[crb + r][j] = mj ? NEGBIG : c[r];
  }

  // ---- image-key scores (cols 272..303), strict causal j < i ----
  for (int jt2 = 0; jt2 < 2; ++jt2) {
    v8f c = vzero8f();
#pragma unroll
    for (int kc = 0; kc < 64; kc += 32) {
      const v16bf a  = frag_a_bf16(Qb + (size_t)(qpos0 + arow) * 64 + kc, kbA);
      const v16bf bb = frag_b(Kb + (size_t)(ibase + jt2 * 16 + bcol) * 64 + kc + kbB);
      c = wmma_bf16(a, bb, c);
    }
    const int jj = jt2 * 16 + bcol;
#pragma unroll
    for (int r = 0; r < 8; ++r) {
      const int ii = i0 + crb + r;
      sc[crb + r][272 + jj] = (jj >= ii) ? NEGBIG : c[r];
    }
  }
  __syncthreads();

  // ---- softmax over 304 real columns ----
  if (lane < 16) {
    float mx = NEGBIG;
    for (int cI = 0; cI < 304; ++cI) mx = fmaxf(mx, sc[lane][cI]);
    float sum = 0.f;
    for (int cI = 0; cI < 304; ++cI) {
      const float e = __expf(sc[lane][cI] - mx);
      sum += e;
      sc[lane][cI] = e;
    }
    const float inv = 1.f / sum;
    for (int cI = 0; cI < 304; ++cI) pr[lane][cI] = (bf16)(sc[lane][cI] * inv);
    for (int cI = 304; cI < LPAD; ++cI) pr[lane][cI] = (bf16)0.f;
  }
  __syncthreads();

  // ---- O = P V  (col<272 -> pos=col; col>=272 -> image pos, clamped pad) ----
  v8f o[4] = {vzero8f(), vzero8f(), vzero8f(), vzero8f()};
  for (int ct = 0; ct < 10; ++ct) {
    const int c0 = ct * 32;
    const v16bf a = frag_a_bf16(&pr[arow][c0], kbA);
#pragma unroll
    for (int s4 = 0; s4 < 4; ++s4) {
      const int d = s4 * 16 + bcol;
      v16bf bb;
      if (c0 + 32 <= 272) {
        bb = frag_b(Vb + (size_t)d * NPAD + c0 + kbB);   // pure text chunk
      } else {
#pragma unroll
        for (int e = 0; e < 16; ++e) {
          const int col = c0 + kbB + e;
          int pos = (col < 272) ? col : (ibase + (col - 272));
          if (pos > NPAD - 1) pos = NPAD - 1;             // padded probs are 0
          bb[e] = Vb[(size_t)d * NPAD + pos];
        }
      }
      o[s4] = wmma_bf16(a, bb, o[s4]);
    }
  }
#pragma unroll
  for (int s4 = 0; s4 < 4; ++s4) {
    const int d = s4 * 16 + bcol;
#pragma unroll
    for (int r = 0; r < 8; ++r) {
      const int pos = qpos0 + crb + r;                    // <= 1280
      Attn[((size_t)b * NPAD + pos) * 512 + h * 64 + d] = (bf16)o[s4][r];
    }
  }
}

// ---------------------------------------------------------------------------
// Kernel 5: output projection, same async-staged 8-wave GEMM structure.
// grid = (8 col-tiles, 11 row-tiles, 32 batches), block = 256.
// ---------------------------------------------------------------------------
__global__ __launch_bounds__(256)
void out_gemm_kernel(const bf16* __restrict__ Attn, const bf16* __restrict__ WoT,
                     const float* __restrict__ bias, float* __restrict__ out) {
  __shared__ __align__(16) bf16 ldsB[2][64 * BSTRIDE];
  const int tid  = threadIdx.x;
  const int wave = tid >> 5;
  const int lane = tid & 31;
  const int b  = blockIdx.z;
  const int mt = blockIdx.y;
  const int nt = blockIdx.x;
  const int row0 = mt * 128 + wave * 16;
  const int col0 = nt * 64;
  const int arow = lane & 15;
  const int kbA  = (lane >> 4) * 8;
  const int bcol = lane & 15;
  const int kbB  = (lane >> 4) * 16;
  const int crb  = (lane >> 4) * 8;

  const int scol = tid >> 2;
  const int sseg = (tid & 3) * 8;
  const bf16* gB = WoT + (size_t)(col0 + scol) * DIMC + sseg;
  bf16* lB0 = &ldsB[0][scol * BSTRIDE + sseg];
  bf16* lB1 = &ldsB[1][scol * BSTRIDE + sseg];

  int grow = row0 + arow;
  if (grow > NPAD - 1) grow = NPAD - 1;         // clamp reads; stores guarded
  const bf16* arowptr = Attn + ((size_t)b * NPAD + grow) * 512;

  v8f acc[4] = {vzero8f(), vzero8f(), vzero8f(), vzero8f()};

  stage16(gB, lB0);
  for (int i = 0; i < DIMC / 32; ++i) {
    const int kc = i * 32;
    if (i + 1 < DIMC / 32) {
      stage16(gB + kc + 32, (i & 1) ? lB0 : lB1);
      async_wait_le1();
    } else {
      async_wait_le0();
    }
    __syncthreads();
    if (kc + 32 < DIMC) __builtin_prefetch(arowptr + kc + 32, 0, 1);
    const v16bf a = frag_a_bf16(arowptr + kc, kbA);
    const bf16* Bbase = &ldsB[i & 1][0];
#pragma unroll
    for (int s = 0; s < 4; ++s) {
      const v16bf bb = frag_b(Bbase + (s * 16 + bcol) * BSTRIDE + kbB);
      acc[s] = wmma_bf16(a, bb, acc[s]);
    }
    __syncthreads();
  }
#pragma unroll
  for (int s = 0; s < 4; ++s) {
    const int col = col0 + s * 16 + bcol;
    const float bv = bias[col];
#pragma unroll
    for (int r = 0; r < 8; ++r) {
      const int pos = row0 + crb + r;
      if (pos < NREAL)
        out[((size_t)b * NREAL + pos) * 512 + col] = acc[s][r] + bv;
    }
  }
}

// ---------------------------------------------------------------------------
extern "C" void kernel_launch(void* const* d_in, const int* in_sizes, int n_in,
                              void* d_out, int out_size, void* d_ws, size_t ws_size,
                              hipStream_t stream) {
  const float*         x     = (const float*)d_in[0];
  const unsigned char* mask  = (const unsigned char*)d_in[1];   // jnp bool -> 1 byte
  const float*         Wqkv  = (const float*)d_in[2];
  const float*         Wout  = (const float*)d_in[3];
  const float*         bout  = (const float*)d_in[4];
  float*               out   = (float*)d_out;

  char* ws = (char*)d_ws;
  size_t off = 0;
  auto alloc = [&](size_t bytes) -> void* {
    void* p = ws + off;
    off += (bytes + 255) & ~(size_t)255;
    return p;
  };
  bf16* WqkvT = (bf16*)alloc((size_t)NQKV * DIMC * 2);       // [1536][512]
  bf16* WoT   = (bf16*)alloc((size_t)DIMC * DIMC * 2);       // [512][512]
  bf16* Qw    = (bf16*)alloc((size_t)256 * NPAD * 64 * 2);   // [bh][1296][64]
  bf16* Kw    = (bf16*)alloc((size_t)256 * NPAD * 64 * 2);
  bf16* Vtw   = (bf16*)alloc((size_t)256 * 64 * NPAD * 2);   // [bh][64][1296]
  bf16* Attn  = (bf16*)alloc((size_t)32 * NPAD * 512 * 2);   // [b][1296][512]
  (void)ws_size; (void)n_in; (void)in_sizes; (void)out_size; // ~172 MB total

  convertT_kernel<<<(DIMC * NQKV + 255) / 256, 256, 0, stream>>>(Wqkv, WqkvT, DIMC, NQKV);
  convertT_kernel<<<(DIMC * DIMC + 255) / 256, 256, 0, stream>>>(Wout, WoT, DIMC, DIMC);

  qkv_gemm_kernel<<<dim3(NQKV / 64, MT128, 32), 256, 0, stream>>>(x, WqkvT, Qw, Kw, Vtw);

  attn_text_kernel<<<dim3(17, 256), 32, 0, stream>>>(Qw, Kw, Vtw, Attn);
  attn_img_kernel<<<dim3(2, IMGW, 256), 32, 0, stream>>>(Qw, Kw, Vtw, mask, Attn);

  out_gemm_kernel<<<dim3(512 / 64, MT128, 32), 256, 0, stream>>>(Attn, WoT, bout, out);
}
